// GCN_66692252172376
// MI455X (gfx1250) — compile-verified, matrix operand order checked by reading, not run on hardware
//
#include <hip/hip_runtime.h>

// ---------------------------------------------------------------------------
// CDNA5 (gfx1250, wave32) WMMA-based implementation of the GNN forward pass.
// All GEMM operand traffic uses float4 (global_load_b128) loads.
// ---------------------------------------------------------------------------

typedef __attribute__((ext_vector_type(16))) _Float16 v16h;
typedef __attribute__((ext_vector_type(8)))  float    v8f;

#define LEAKY_NEG 0.1f

__device__ __forceinline__ v16h pack16(const float4 a0, const float4 a1,
                                       const float4 a2, const float4 a3,
                                       const float s) {
  v16h v;
  v[0]  = (_Float16)(a0.x * s); v[1]  = (_Float16)(a0.y * s);
  v[2]  = (_Float16)(a0.z * s); v[3]  = (_Float16)(a0.w * s);
  v[4]  = (_Float16)(a1.x * s); v[5]  = (_Float16)(a1.y * s);
  v[6]  = (_Float16)(a1.z * s); v[7]  = (_Float16)(a1.w * s);
  v[8]  = (_Float16)(a2.x * s); v[9]  = (_Float16)(a2.y * s);
  v[10] = (_Float16)(a2.z * s); v[11] = (_Float16)(a2.w * s);
  v[12] = (_Float16)(a3.x * s); v[13] = (_Float16)(a3.y * s);
  v[14] = (_Float16)(a3.z * s); v[15] = (_Float16)(a3.w * s);
  return v;
}

// ---------------------------------------------------------------------------
// Generic WMMA GEMM:  C[M,Nout] = act( Aeff[M,K] * B[K,Nout] + bias )
//   Aeff row m:  amode 0: A1[m,:] (stride lda)
//                amode 1: concat(A1[m, :ka1], A2[m, :K-ka1])
//                amode 2: concat(A1[idx1[m], :ka1], A1[idx2[m], :ka1])  (edge gather)
//                amode 3: concat(A1[tok(m), :ka1], A2[m, :K-ka1]),
//                         tok(m) = idx1[m*tokStride + tokCol]           (LSTM input)
//   Optional per-row scale rowscale[m] on Aeff (GCN degree norm).
//   B(n,k) = k<kb1 ? B1[n*ldb1+k] : B2[n*ldb2+(k-kb1)]   (weights as [out,in])
// One wave computes a 16(M) x 64(N) strip via 4 accumulators.
// Requires: M%16==0, Nout%64==0, K%32==0, ka1%8==0, kb1%8==0, strides%4==0.
// ---------------------------------------------------------------------------
__global__ void gemm_wmma_f16(
    const float* __restrict__ A1, const float* __restrict__ A2,
    const int* __restrict__ idx1, const int* __restrict__ idx2,
    int tokStride, int tokCol, int amode, int lda, int ka1,
    const float* __restrict__ rowscale,
    const float* __restrict__ B1, const float* __restrict__ B2,
    int ldb1, int ldb2, int kb1,
    const float* __restrict__ bias,
    float* __restrict__ C, int ldC,
    int M, int Nout, int K, int act)
{
  const int lane = threadIdx.x & 31;
  const int wave = threadIdx.x >> 5;
  const int nTilesN = Nout >> 6;
  const int units = (M >> 4) * nTilesN;
  const int unit = blockIdx.x * (blockDim.x >> 5) + wave;
  if (unit >= units) return;                 // wave-uniform exit
  const int tm = unit / nTilesN;
  const int tn = unit - tm * nTilesN;
  const int m0 = tm << 4;
  const int n0 = tn << 6;
  const int l15 = lane & 15;
  const int lh  = lane >> 4;
  const int mRow = m0 + l15;

  // A row pointers (lanes 0-15 / 16-31 cover the same rows, different K-chunks)
  const float* arow1;
  const float* arow2 = nullptr;
  if (amode == 0) {
    arow1 = A1 + (size_t)mRow * lda;
    arow2 = arow1;
  } else if (amode == 1) {
    arow1 = A1 + (size_t)mRow * ka1;
    arow2 = A2 + (size_t)mRow * (K - ka1);
  } else if (amode == 2) {
    arow1 = A1 + (size_t)idx1[mRow] * ka1;
    arow2 = A1 + (size_t)idx2[mRow] * ka1;
  } else {
    const int tok = idx1[(size_t)mRow * tokStride + tokCol];
    arow1 = A1 + (size_t)tok * ka1;
    arow2 = A2 + (size_t)mRow * (K - ka1);
  }
  const float rs = rowscale ? rowscale[mRow] : 1.0f;

  // B row pointers for the 4 N-subtiles (lane l15 = column n)
  const float* brow1[4];
  const float* brow2[4];
  #pragma unroll
  for (int j = 0; j < 4; ++j) {
    const int n = n0 + j * 16 + l15;
    brow1[j] = B1 + (size_t)n * ldb1;
    brow2[j] = B2 ? (B2 + (size_t)n * ldb2) : brow1[j];
  }

  v8f acc[4] = {};

  for (int k0 = 0; k0 < K; k0 += 32) {
    const int kb = k0 + lh * 8;              // K-chunk base for this half-wave
    const int kc = kb + 16;
    // A: two contiguous 8-float runs -> 4x float4
    const float* pa0 = (kb < ka1) ? (arow1 + kb) : (arow2 + (kb - ka1));
    const float* pa1 = (kc < ka1) ? (arow1 + kc) : (arow2 + (kc - ka1));
    const float4 a0 = *(const float4*)(pa0);
    const float4 a1 = *(const float4*)(pa0 + 4);
    const float4 a2 = *(const float4*)(pa1);
    const float4 a3 = *(const float4*)(pa1 + 4);
    const v16h av = pack16(a0, a1, a2, a3, rs);
    #pragma unroll
    for (int j = 0; j < 4; ++j) {
      const float* pb0 = (kb < kb1) ? (brow1[j] + kb) : (brow2[j] + (kb - kb1));
      const float* pb1 = (kc < kb1) ? (brow1[j] + kc) : (brow2[j] + (kc - kb1));
      const float4 b0 = *(const float4*)(pb0);
      const float4 b1 = *(const float4*)(pb0 + 4);
      const float4 b2 = *(const float4*)(pb1);
      const float4 b3 = *(const float4*)(pb1 + 4);
      const v16h bv = pack16(b0, b1, b2, b3, 1.0f);
      acc[j] = __builtin_amdgcn_wmma_f32_16x16x32_f16(
          false, av, false, bv, (short)0, acc[j], false, false);
    }
  }

  // Epilogue: C element r of acc[j] is (M = m0 + r + 8*lh, N = n0 + 16j + l15)
  #pragma unroll
  for (int j = 0; j < 4; ++j) {
    const int col = n0 + j * 16 + l15;
    const float bvl = bias ? bias[col] : 0.0f;
    #pragma unroll
    for (int r = 0; r < 8; ++r) {
      const int row = m0 + r + lh * 8;
      float v = acc[j][r] + bvl;
      if (act == 1) v = (v >= 0.0f) ? v : LEAKY_NEG * v;
      C[(size_t)row * ldC + col] = v;
    }
  }
}

// ---------------------------------------------------------------------------
// Elementwise / scatter kernels
// ---------------------------------------------------------------------------
__global__ void zero_f32v4(float4* __restrict__ p, size_t n4) {
  size_t t = (size_t)blockIdx.x * blockDim.x + threadIdx.x;
  if (t < n4) p[t] = make_float4(0.f, 0.f, 0.f, 0.f);
}

// Wt[n*K + k] = W[k*Nout + n]   (GCN weights [in,out] -> [out,in])
__global__ void transpose_w(const float* __restrict__ src, float* __restrict__ dst,
                            int K, int Nout) {
  int t = blockIdx.x * blockDim.x + threadIdx.x;
  if (t >= K * Nout) return;
  int k = t / Nout, n = t - k * Nout;
  dst[(size_t)n * K + k] = src[t];
}

__device__ __forceinline__ float sigm(float x) { return 1.0f / (1.0f + expf(-x)); }

// LSTM cell update, 4 channels per thread: g[N,512] (i|f|g|o), h/c [N,128].
__global__ void lstm_update(const float* __restrict__ g,
                            float* __restrict__ h, float* __restrict__ c,
                            int n) {
  int t = blockIdx.x * blockDim.x + threadIdx.x;
  if (t >= n * 32) return;
  int row = t >> 5, j = (t & 31) * 4;
  const float* gr = g + (size_t)row * 512;
  const float4 gi = *(const float4*)(gr + j);
  const float4 gf = *(const float4*)(gr + 128 + j);
  const float4 gg = *(const float4*)(gr + 256 + j);
  const float4 go = *(const float4*)(gr + 384 + j);
  float* cp = c + (size_t)row * 128 + j;
  float* hp = h + (size_t)row * 128 + j;
  const float4 cv = *(const float4*)cp;
  float4 c2, h2;
  c2.x = sigm(gf.x) * cv.x + sigm(gi.x) * tanhf(gg.x);
  c2.y = sigm(gf.y) * cv.y + sigm(gi.y) * tanhf(gg.y);
  c2.z = sigm(gf.z) * cv.z + sigm(gi.z) * tanhf(gg.z);
  c2.w = sigm(gf.w) * cv.w + sigm(gi.w) * tanhf(gg.w);
  h2.x = sigm(go.x) * tanhf(c2.x);
  h2.y = sigm(go.y) * tanhf(c2.y);
  h2.z = sigm(go.z) * tanhf(c2.z);
  h2.w = sigm(go.w) * tanhf(c2.w);
  *(float4*)cp = c2;
  *(float4*)hp = h2;
}

// h0 columns 128..543 (float4 groups): cat_emb | co_emb | sl_emb | ip
__global__ void fill_feat(const int* __restrict__ ic, const int* __restrict__ ico,
                          const int* __restrict__ isl, const float* __restrict__ ip,
                          const float* __restrict__ cat_emb, const float* __restrict__ co_emb,
                          const float* __restrict__ sl_emb, float* __restrict__ h0,
                          int n) {
  int t = blockIdx.x * blockDim.x + threadIdx.x;
  if (t >= n * 104) return;
  int row = t / 104, j = (t - row * 104) * 4;
  float4 v;
  if (j < 128)      v = *(const float4*)(cat_emb + (size_t)ic[row]  * 128 + j);
  else if (j < 256) v = *(const float4*)(co_emb  + (size_t)ico[row] * 128 + (j - 128));
  else if (j < 384) v = *(const float4*)(sl_emb  + (size_t)isl[row] * 128 + (j - 256));
  else              v = *(const float4*)(ip      + (size_t)row      * 32  + (j - 384));
  *(float4*)(h0 + (size_t)row * 544 + 128 + j) = v;
}

__global__ void degree_count(const int* __restrict__ idx, float* __restrict__ deg, int n) {
  int t = blockIdx.x * blockDim.x + threadIdx.x;
  if (t < n) atomicAdd(&deg[idx[t]], 1.0f);
}

__global__ void rsqrt_max1(float* __restrict__ d, int n) {
  int t = blockIdx.x * blockDim.x + threadIdx.x;
  if (t < n) d[t] = rsqrtf(fmaxf(d[t], 1.0f));
}

// agg[dst[e], :] += m[src[e], :]  — 4 columns per thread, float4 gather.
__global__ void scatter_add(const float* __restrict__ m, const int* __restrict__ src,
                            const int* __restrict__ dst, float* __restrict__ agg,
                            int nEdges) {
  int t = blockIdx.x * blockDim.x + threadIdx.x;
  int e = t >> 6;                      // 64 threads per edge, 4 cols each (256 cols)
  if (e >= nEdges) return;
  int c4 = (t & 63) * 4;
  const float4 v = *(const float4*)(m + (size_t)src[e] * 256 + c4);
  float* dp = agg + (size_t)dst[e] * 256 + c4;
  atomicAdd(dp + 0, v.x);
  atomicAdd(dp + 1, v.y);
  atomicAdd(dp + 2, v.z);
  atomicAdd(dp + 3, v.w);
}

// out = (mode? out : 0) + agg*nd[row] + bias[col]; optional leaky after. 256 cols.
__global__ void gcn_combine(const float* __restrict__ agg, const float* __restrict__ nd,
                            const float* __restrict__ bias, float* __restrict__ out,
                            int n, int mode, int leaky) {
  int t = blockIdx.x * blockDim.x + threadIdx.x;
  if (t >= n * 64) return;
  int row = t >> 6, c4 = (t & 63) * 4;
  const float s = nd[row];
  const float4 a = *(const float4*)(agg + (size_t)row * 256 + c4);
  const float4 b = *(const float4*)(bias + c4);
  float4 v = make_float4(a.x * s + b.x, a.y * s + b.y, a.z * s + b.z, a.w * s + b.w);
  float4* op = (float4*)(out + (size_t)row * 256 + c4);
  if (mode) { const float4 o = *op; v.x += o.x; v.y += o.y; v.z += o.z; v.w += o.w; }
  if (leaky) {
    v.x = (v.x >= 0.f) ? v.x : LEAKY_NEG * v.x;
    v.y = (v.y >= 0.f) ? v.y : LEAKY_NEG * v.y;
    v.z = (v.z >= 0.f) ? v.z : LEAKY_NEG * v.z;
    v.w = (v.w >= 0.f) ? v.w : LEAKY_NEG * v.w;
  }
  *op = v;
}

// Column sums / sums of squares over x[rows,256]; one block handles 256 rows.
__global__ void col_stats(const float* __restrict__ x, float* __restrict__ sum,
                          float* __restrict__ sumsq, int rows) {
  int c = threadIdx.x;               // blockDim.x == 256 == cols
  int r0 = blockIdx.x * 256;
  int r1 = min(r0 + 256, rows);
  float s = 0.0f, q = 0.0f;
  for (int r = r0; r < r1; ++r) {
    float v = x[(size_t)r * 256 + c];
    s += v; q += v * v;
  }
  atomicAdd(&sum[c], s);
  atomicAdd(&sumsq[c], q);
}

__global__ void bn_finalize(const float* __restrict__ sum, const float* __restrict__ sumsq,
                            float* __restrict__ mean, float* __restrict__ inv, int rows) {
  int c = threadIdx.x;
  if (c < 256) {
    float mu = sum[c] / (float)rows;
    float var = sumsq[c] / (float)rows - mu * mu;
    mean[c] = mu;
    inv[c] = rsqrtf(var + 1e-5f);
  }
}

// BN-apply + ReLU + [256 -> 2] matmul, one wave32 per row.
__global__ void classify(const float* __restrict__ x, const float* __restrict__ mean,
                         const float* __restrict__ inv, const float* __restrict__ gamma,
                         const float* __restrict__ beta, const float* __restrict__ W2,
                         const float* __restrict__ b2, float* __restrict__ out, int rows) {
  int warp = (blockIdx.x * blockDim.x + threadIdx.x) >> 5;
  int lane = threadIdx.x & 31;
  if (warp >= rows) return;
  const float* xr = x + (size_t)warp * 256;
  float d0 = 0.0f, d1 = 0.0f;
  for (int c = lane; c < 256; c += 32) {
    float v = (xr[c] - mean[c]) * inv[c] * gamma[c] + beta[c];
    v = fmaxf(v, 0.0f);
    d0 += v * W2[c];
    d1 += v * W2[256 + c];
  }
  for (int o = 16; o > 0; o >>= 1) {
    d0 += __shfl_down(d0, o, 32);
    d1 += __shfl_down(d1, o, 32);
  }
  if (lane == 0) {
    out[(size_t)warp * 2 + 0] = d0 + b2[0];
    out[(size_t)warp * 2 + 1] = d1 + b2[1];
  }
}

// ---------------------------------------------------------------------------
// Host orchestration
// ---------------------------------------------------------------------------
static inline int cdiv(long long a, int b) { return (int)((a + b - 1) / b); }

static inline void zero_f32(hipStream_t s, float* p, size_t n) {
  zero_f32v4<<<cdiv((long long)(n / 4), 256), 256, 0, s>>>((float4*)p, n / 4);
}

static inline void launch_gemm(hipStream_t s, int M, int Nout, int K,
    const float* A1, const float* A2, const int* idx1, const int* idx2,
    int tokStride, int tokCol, int amode, int lda, int ka1, const float* rowscale,
    const float* B1, const float* B2, int ldb1, int ldb2, int kb1,
    const float* bias, float* C, int ldC, int act) {
  int units = (M / 16) * (Nout / 64);
  int blocks = (units + 7) / 8;        // 8 waves (256 threads) per block
  gemm_wmma_f16<<<blocks, 256, 0, s>>>(A1, A2, idx1, idx2, tokStride, tokCol,
      amode, lda, ka1, rowscale, B1, B2, ldb1, ldb2, kb1, bias, C, ldC, M, Nout, K, act);
}

extern "C" void kernel_launch(void* const* d_in, const int* in_sizes, int n_in,
                              void* d_out, int out_size, void* d_ws, size_t ws_size,
                              hipStream_t stream) {
  // --- params (JAX pytree sorted-key flatten order) ---
  const float* W1      = (const float*)d_in[0];   // [256,512]
  const float* W2      = (const float*)d_in[1];   // [2,256]
  const float* W_fc    = (const float*)d_in[2];   // [128,256]
  const float* Whh_b   = (const float*)d_in[3];   // [512,128]
  const float* Whh_f   = (const float*)d_in[4];
  const float* Wih_b   = (const float*)d_in[5];
  const float* Wih_f   = (const float*)d_in[6];
  const float* b1      = (const float*)d_in[7];
  const float* b2      = (const float*)d_in[8];
  const float* b_b     = (const float*)d_in[9];
  const float* b_f     = (const float*)d_in[10];
  const float* b_fc    = (const float*)d_in[11];
  const float* beta    = (const float*)d_in[12];
  const float* cat_emb = (const float*)d_in[13];
  const float* co_emb  = (const float*)d_in[14];
  const float* gamma   = (const float*)d_in[15];
  const float* g0_Wsim = (const float*)d_in[16];  // [544,256]
  const float* g0_Wusr = (const float*)d_in[17];
  const float* g0_bsim = (const float*)d_in[18];
  const float* g0_busr = (const float*)d_in[19];
  const float* g1_Wsim = (const float*)d_in[20];  // [256,256]
  const float* g1_Wusr = (const float*)d_in[21];
  const float* g1_bsim = (const float*)d_in[22];
  const float* g1_busr = (const float*)d_in[23];
  const float* sl_emb  = (const float*)d_in[24];
  const float* url_emb = (const float*)d_in[25];  // [128,128]
  const int*   in_s    = (const int*)d_in[26];    // [N,16]
  const int*   in_c    = (const int*)d_in[28];
  const int*   in_co   = (const int*)d_in[29];
  const int*   in_sl   = (const int*)d_in[30];
  const float* in_ip   = (const float*)d_in[31];
  const int*   sim_src = (const int*)d_in[32];
  const int*   sim_dst = (const int*)d_in[33];
  const int*   usr_src = (const int*)d_in[34];
  const int*   usr_dst = (const int*)d_in[35];
  const int*   es_src  = (const int*)d_in[36];
  const int*   es_dst  = (const int*)d_in[37];

  const int N    = in_sizes[28];                  // 20000
  const int L    = in_sizes[26] / N;              // 16
  const int E    = in_sizes[32];                  // 300000
  const int ECLS = in_sizes[36];                  // 100000
  float* out = (float*)d_out;

  // --- workspace layout (floats). x overlays the LSTM/h0 region (dead by then).
  float* W = (float*)d_ws;
  size_t off = 0;
  float* hf = W + off; off += (size_t)N * 128;
  float* cf = W + off; off += (size_t)N * 128;
  float* hb = W + off; off += (size_t)N * 128;
  float* cb = W + off; off += (size_t)N * 128;
  float* g  = W + off; off += (size_t)N * 512;
  float* h0 = W + off; off += (size_t)N * 544;    // end of overlay region
  float* x  = (float*)d_ws;                        // [ECLS,256], overlays hf..h0
  float* norms = W + off; off += (size_t)4 * N;   // ns_sim|nd_sim|ns_usr|nd_usr
  float* mbuf  = W + off; off += (size_t)N * 256;
  float* agg   = W + off; off += (size_t)N * 256;
  float* h1    = W + off; off += (size_t)N * 256;
  float* h2    = W + off; off += (size_t)N * 256;
  float* ssum  = W + off; off += 256;
  float* ssq   = W + off; off += 256;
  float* smean = W + off; off += 256;
  float* sinv  = W + off; off += 256;
  float* wt0s  = W + off; off += (size_t)544 * 256;  // transposed GCN weights
  float* wt0u  = W + off; off += (size_t)544 * 256;
  float* wt1s  = W + off; off += (size_t)256 * 256;
  float* wt1u  = W + off; off += (size_t)256 * 256;
  float* ns_sim = norms, *nd_sim = norms + N, *ns_usr = norms + 2 * N, *nd_usr = norms + 3 * N;

  // --- 0. transpose GCN weights [in,out] -> [out,in] so GEMM B loads are b128
  transpose_w<<<cdiv(544 * 256, 256), 256, 0, stream>>>(g0_Wsim, wt0s, 544, 256);
  transpose_w<<<cdiv(544 * 256, 256), 256, 0, stream>>>(g0_Wusr, wt0u, 544, 256);
  transpose_w<<<cdiv(256 * 256, 256), 256, 0, stream>>>(g1_Wsim, wt1s, 256, 256);
  transpose_w<<<cdiv(256 * 256, 256), 256, 0, stream>>>(g1_Wusr, wt1u, 256, 256);

  // --- 1. init LSTM state (hf|cf|hb|cb contiguous)
  zero_f32(stream, hf, (size_t)N * 512);

  // --- 2. BiLSTM: g = [emb_t | h] @ [Wih ; Whh].T + b, then cell update
  for (int t = 0; t < L; ++t) {
    launch_gemm(stream, N, 512, 256, url_emb, hf, in_s, nullptr, L, t, /*amode*/3,
                0, 128, nullptr, Wih_f, Whh_f, 128, 128, 128, b_f, g, 512, 0);
    lstm_update<<<cdiv((long long)N * 32, 256), 256, 0, stream>>>(g, hf, cf, N);
  }
  for (int t = 0; t < L; ++t) {
    launch_gemm(stream, N, 512, 256, url_emb, hb, in_s, nullptr, L, L - 1 - t, 3,
                0, 128, nullptr, Wih_b, Whh_b, 128, 128, 128, b_b, g, 512, 0);
    lstm_update<<<cdiv((long long)N * 32, 256), 256, 0, stream>>>(g, hb, cb, N);
  }

  // --- 3. h_url = leaky([hf|hb] @ W_fc.T + b_fc) -> h0[:, 0:128]
  launch_gemm(stream, N, 128, 256, hf, hb, nullptr, nullptr, 0, 0, /*amode*/1,
              0, 128, nullptr, W_fc, nullptr, 256, 0, 256, b_fc, h0, 544, 1);
  // h0[:, 128:544] = cat|co|sl|ip
  fill_feat<<<cdiv((long long)N * 104, 256), 256, 0, stream>>>(
      in_c, in_co, in_sl, in_ip, cat_emb, co_emb, sl_emb, h0, N);

  // --- 4. degree norms: rsqrt(max(deg,1))
  zero_f32(stream, norms, (size_t)4 * N);
  degree_count<<<cdiv(E, 256), 256, 0, stream>>>(sim_src, ns_sim, E);
  degree_count<<<cdiv(E, 256), 256, 0, stream>>>(sim_dst, nd_sim, E);
  degree_count<<<cdiv(E, 256), 256, 0, stream>>>(usr_src, ns_usr, E);
  degree_count<<<cdiv(E, 256), 256, 0, stream>>>(usr_dst, nd_usr, E);
  rsqrt_max1<<<cdiv(4 * N, 256), 256, 0, stream>>>(norms, 4 * N);

  // --- 5. GCN layers: out = leaky( sum_rel( scatter((h*ns)@W)[dst]*nd + b ) )
  const float* hin = h0; int kin = 544;
  const float* Wr[2][2] = {{wt0s, wt0u}, {wt1s, wt1u}};
  const float* br[2][2] = {{g0_bsim, g0_busr}, {g1_bsim, g1_busr}};
  float* hout[2] = {h1, h2};
  const int* esrc[2] = {sim_src, usr_src};
  const int* edst[2] = {sim_dst, usr_dst};
  const float* ens[2] = {ns_sim, ns_usr};
  const float* end_[2] = {nd_sim, nd_usr};
  for (int layer = 0; layer < 2; ++layer) {
    for (int rel = 0; rel < 2; ++rel) {
      launch_gemm(stream, N, 256, kin, hin, nullptr, nullptr, nullptr, 0, 0, /*amode*/0,
                  kin, kin, ens[rel], Wr[layer][rel], nullptr, kin, 0, kin,
                  nullptr, mbuf, 256, 0);
      zero_f32(stream, agg, (size_t)N * 256);
      scatter_add<<<cdiv((long long)E * 64, 256), 256, 0, stream>>>(
          mbuf, esrc[rel], edst[rel], agg, E);
      gcn_combine<<<cdiv((long long)N * 64, 256), 256, 0, stream>>>(
          agg, end_[rel], br[layer][rel], hout[layer], N, /*mode*/rel, /*leaky*/rel);
    }
    hin = hout[layer];
    kin = 256;
  }

  // --- 6. classifier: x = [h2[src]|h2[dst]] @ W1.T + b1   (h0/g dead -> overlay)
  launch_gemm(stream, ECLS, 256, 512, h2, nullptr, es_src, es_dst, 0, 0, /*amode*/2,
              0, 256, nullptr, W1, nullptr, 512, 0, 512, b1, x, 256, 0);

  // --- 7. batchnorm (batch stats) + relu + W2 -> out[ECLS,2]
  zero_f32(stream, ssum, 512);
  col_stats<<<cdiv(ECLS, 256), 256, 0, stream>>>(x, ssum, ssq, ECLS);
  bn_finalize<<<1, 256, 0, stream>>>(ssum, ssq, smean, sinv, ECLS);
  classify<<<cdiv((long long)ECLS * 32, 256), 256, 0, stream>>>(
      x, smean, sinv, gamma, beta, W2, b2, out, ECLS);

  (void)n_in; (void)out_size; (void)ws_size;
}
